// FourrierKernelAttention_84035330113828
// MI455X (gfx1250) — matrix operand order repaired and checked by reading, not compiled, not run on hardware
//
#include <hip/hip_runtime.h>
#include <math.h>

typedef __attribute__((ext_vector_type(16))) _Float16 v16h;
typedef __attribute__((ext_vector_type(8)))  _Float16 v8h;
typedef __attribute__((ext_vector_type(8)))  float    v8f;

union V16H { v16h v; v8h half8[2]; _Float16 h[16]; };
union V8H  { v8h  v; _Float16 h[8]; };

#define WMMA(A, B, C) __builtin_amdgcn_wmma_f32_16x16x32_f16( \
    false, (A), false, (B), (short)0, (C), false, false)

// A-fragment K mapping for 16x32 f16 A tiles:
// lanes 0-15 (hi=0): elems 0..7 -> K 0..7,  elems 8..15 -> K 16..23
// lanes 16-31(hi=1): elems 0..7 -> K 8..15, elems 8..15 -> K 24..31
__device__ __forceinline__ int a_kmap(int e, int hi) {
  return (e < 8) ? (8 * hi + e) : (8 + 8 * hi + e);
}

__device__ __forceinline__ float softplus_fast(float x) {
  return (x > 20.f) ? x : __logf(1.f + __expf(x));
}

// ---------------------------------------------------------------------------
// Kernel 1: fused projections  Q = softplus(query@Wq^T), K = softplus(key@Wk^T),
// V = key@Wv^T, written to workspace in (N,H,L,64) fp32 layout.
// One 16x16 output tile per wave; K-loop of 16 WMMAs (K=512).
// ---------------------------------------------------------------------------
__global__ __launch_bounds__(128) void fka_proj_kernel(
    const float* __restrict__ query, const float* __restrict__ key,
    const float* __restrict__ Wq, const float* __restrict__ Wk,
    const float* __restrict__ Wv,
    float* __restrict__ Qo, float* __restrict__ Ko, float* __restrict__ Vo) {
  const int DIM = 512;
  const int lane = threadIdx.x & 31;
  const int wave = threadIdx.x >> 5;
  const int g = blockIdx.x * 4 + wave;          // 0 .. 12287 tile id
  const int TN = DIM / 16;                      // 32
  const int TM = 2048 / 16;                     // 128
  const int which = g / (TM * TN);              // 0:Q 1:K 2:V
  const int rem = g % (TM * TN);
  const int mt = rem / TN, nt = rem % TN;

  const float* X = (which == 0) ? query : key;
  const float* W = (which == 0) ? Wq : (which == 1 ? Wk : Wv);
  float* Out = (which == 0) ? Qo : (which == 1 ? Ko : Vo);

  const int hi = lane >> 4;
  const int mloc = lane & 15;
  const int mrow = mt * 16 + mloc;

  v8f acc = {};
  for (int kb = 0; kb < DIM; kb += 32) {
    V16H a, b;
    // A rows are contiguous 8-float runs in global memory -> float4 pairs
    const float4* xa0 = (const float4*)(X + mrow * DIM + kb + 8 * hi);
    const float4* xa1 = (const float4*)(X + mrow * DIM + kb + 16 + 8 * hi);
#pragma unroll
    for (int q4 = 0; q4 < 2; ++q4) {
      const float4 lo = xa0[q4], hi4 = xa1[q4];
      a.h[4 * q4 + 0] = (_Float16)lo.x;  a.h[4 * q4 + 1] = (_Float16)lo.y;
      a.h[4 * q4 + 2] = (_Float16)lo.z;  a.h[4 * q4 + 3] = (_Float16)lo.w;
      a.h[8 + 4 * q4 + 0] = (_Float16)hi4.x; a.h[8 + 4 * q4 + 1] = (_Float16)hi4.y;
      a.h[8 + 4 * q4 + 2] = (_Float16)hi4.z; a.h[8 + 4 * q4 + 3] = (_Float16)hi4.w;
    }
#pragma unroll
    for (int e = 0; e < 16; ++e) {
      // B[k][o] = W[o][k]  (X @ W^T); lane%16 = N col, e -> K via a_kmap
      b.h[e] = (_Float16)W[(nt * 16 + mloc) * DIM + (kb + a_kmap(e, hi))];
    }
    acc = WMMA(a.v, b.v, acc);
  }

#pragma unroll
  for (int j = 0; j < 8; ++j) {
    const int m = mt * 16 + j + 8 * hi;
    const int o = nt * 16 + mloc;
    float x = acc[j];
    if (which < 2) x = softplus_fast(x);
    const int n = m >> 10, l = m & 1023;
    const int h = o >> 6, d = o & 63;
    Out[(((size_t)(n * 8 + h) * 1024) + l) * 64 + d] = x;
  }
}

// ---------------------------------------------------------------------------
// Kernel 2: chunked causal linear attention with 3-branch RPE trig split.
// One block per (n,h); 4 waves; chunk C=16, 64 chunks sequential.
// States S_b (64x64 fp32, b=cos/sin/plain) live in LDS; wave w owns columns
// [16w,16w+16) of every S_b (exclusive read-modify-write, no cross-wave dep).
// Intra-chunk scores are branch-summed BEFORE P@V (trig identity), so only
// one masked P@V per chunk.
// ---------------------------------------------------------------------------
__global__ __launch_bounds__(128) void fka_attn_kernel(
    const float* __restrict__ Q, const float* __restrict__ K,
    const float* __restrict__ V, const float* __restrict__ pos_w,
    const float* __restrict__ pos_b, float* __restrict__ out) {
  extern __shared__ char smem[];
  float* S    = (float*)smem;            // [3][64][64]
  float* zf   = S + 3 * 64 * 64;         // [3][64] running sum of kf
  float* pw   = zf + 3 * 64;             // [64]
  float* pbs  = pw + 64;                 // [64] sigmoid(bias)*pi
  float* den  = pbs + 64;                // [16]
  float* denp = den + 16;                // [16][8]
  _Float16* qfh = (_Float16*)(denp + 16 * 8);  // [3][16][64]
  _Float16* kfh = qfh + 3 * 16 * 64;           // [3][16][64]
  _Float16* vh  = kfh + 3 * 16 * 64;           // [16][64]
  _Float16* pt  = vh + 16 * 64;                // [16][16] masked scores

  const int t = threadIdx.x;
  const int lane = t & 31, wave = t >> 5;
  const int hi = lane >> 4, mloc = lane & 15;
  const int nh = blockIdx.x;            // 0..15
  const int n = nh >> 3, h = nh & 7;

  for (int i = t; i < 3 * 64 * 64; i += 128) S[i] = 0.f;
  for (int i = t; i < 3 * 64; i += 128) zf[i] = 0.f;
  if (t < 64) {
    pw[t] = pos_w[h * 64 + t];  // P==1
    const float sg = 1.f / (1.f + __expf(-pos_b[h * 64 + t]));
    pbs[t] = sg * 3.14159265358979323846f;
  }
  __syncthreads();

  const float* Qb = Q + (size_t)nh * 1024 * 64;
  const float* Kb = K + (size_t)nh * 1024 * 64;
  const float* Vb = V + (size_t)nh * 1024 * 64;

  for (int c = 0; c < 64; ++c) {
    const int l0 = c * 16;

    // ---- phase 1: feature generation. thread t -> row l = t/8, 8 contiguous
    //      d starting at dg = (t%8)*8; float4-pair loads, v8h stores. ----
    {
      const int l = t >> 3, dg = (t & 7) * 8;
      const size_t base = (size_t)(l0 + l) * 64 + dg;
      const float4* q4 = (const float4*)(Qb + base);
      const float4* k4 = (const float4*)(Kb + base);
      const float4* v4 = (const float4*)(Vb + base);
      float qv[8], kv[8], vv[8];
#pragma unroll
      for (int p = 0; p < 2; ++p) {
        const float4 a = q4[p], b = k4[p], cfl = v4[p];
        qv[4 * p + 0] = a.x; qv[4 * p + 1] = a.y; qv[4 * p + 2] = a.z; qv[4 * p + 3] = a.w;
        kv[4 * p + 0] = b.x; kv[4 * p + 1] = b.y; kv[4 * p + 2] = b.z; kv[4 * p + 3] = b.w;
        vv[4 * p + 0] = cfl.x; vv[4 * p + 1] = cfl.y; vv[4 * p + 2] = cfl.z; vv[4 * p + 3] = cfl.w;
      }
      V8H qc, qs, qp, kc, ks, kp, vz;
#pragma unroll
      for (int e = 0; e < 8; ++e) {
        const int d = dg + e;
        const float ang = (float)(l0 + l) * pw[d];
        const float aq = ang + pbs[d];
        qc.h[e] = (_Float16)(qv[e] * __cosf(aq));
        qs.h[e] = (_Float16)(qv[e] * __sinf(aq));
        qp.h[e] = (_Float16)qv[e];
        kc.h[e] = (_Float16)(kv[e] * __cosf(ang));
        ks.h[e] = (_Float16)(kv[e] * __sinf(ang));
        kp.h[e] = (_Float16)kv[e];
        vz.h[e] = (_Float16)vv[e];
      }
      const int o = l * 64 + dg;
      *(v8h*)(qfh + 0 * 1024 + o) = qc.v;
      *(v8h*)(qfh + 1 * 1024 + o) = qs.v;
      *(v8h*)(qfh + 2 * 1024 + o) = qp.v;
      *(v8h*)(kfh + 0 * 1024 + o) = kc.v;
      *(v8h*)(kfh + 1 * 1024 + o) = ks.v;
      *(v8h*)(kfh + 2 * 1024 + o) = kp.v;
      *(v8h*)(vh + o) = vz.v;
    }
    __syncthreads();

    // ---- phase 2a: inter-chunk numerator (all waves); scores (wave 0) ----
    v8f acc = {};
#pragma unroll
    for (int b = 0; b < 3; ++b) {
      V16H a0, a1, s0, s1;
      // qf A-fragments: contiguous 8-half runs -> vector LDS loads
      a0.half8[0] = *(const v8h*)(qfh + b * 1024 + mloc * 64 + 8 * hi);
      a0.half8[1] = *(const v8h*)(qfh + b * 1024 + mloc * 64 + 16 + 8 * hi);
      a1.half8[0] = *(const v8h*)(qfh + b * 1024 + mloc * 64 + 32 + 8 * hi);
      a1.half8[1] = *(const v8h*)(qfh + b * 1024 + mloc * 64 + 48 + 8 * hi);
#pragma unroll
      for (int e = 0; e < 16; ++e) {
        // B[k=d][n=col] = S_b[d][16w+n]; lane%16 = n, e -> d via a_kmap
        s0.h[e] = (_Float16)S[(b * 64 + a_kmap(e, hi)) * 64 + wave * 16 + mloc];
        s1.h[e] = (_Float16)S[(b * 64 + 32 + a_kmap(e, hi)) * 64 + wave * 16 + mloc];
      }
      acc = WMMA(a0.v, s0.v, acc);
      acc = WMMA(a1.v, s1.v, acc);
    }

    if (wave == 0) {  // wave-uniform branch: EXEC all-ones within the wave
      v8f p = {};
#pragma unroll
      for (int b = 0; b < 3; ++b) {
        V16H a0, a1, k0, k1;
        a0.half8[0] = *(const v8h*)(qfh + b * 1024 + mloc * 64 + 8 * hi);
        a0.half8[1] = *(const v8h*)(qfh + b * 1024 + mloc * 64 + 16 + 8 * hi);
        a1.half8[0] = *(const v8h*)(qfh + b * 1024 + mloc * 64 + 32 + 8 * hi);
        a1.half8[1] = *(const v8h*)(qfh + b * 1024 + mloc * 64 + 48 + 8 * hi);
#pragma unroll
        for (int e = 0; e < 16; ++e) {
          // B[k=d][n=l'] = kf[l'][d]; lane%16 = l', e -> d via a_kmap
          k0.h[e] = kfh[b * 1024 + mloc * 64 + a_kmap(e, hi)];
          k1.h[e] = kfh[b * 1024 + mloc * 64 + 32 + a_kmap(e, hi)];
        }
        p = WMMA(a0.v, k0.v, p);
        p = WMMA(a1.v, k1.v, p);
      }
      // causal mask (key index <= query index), store as f16 for A-operand use
#pragma unroll
      for (int j = 0; j < 8; ++j) {
        const int M = j + 8 * hi;                  // query row in chunk
        pt[M * 16 + mloc] = (_Float16)((mloc <= M) ? p[j] : 0.f);
      }
    }
    __syncthreads();

    // ---- phase 2b: intra-chunk P@V, then state update S_b += kf^T @ v ----
    {
      V16H pa, bv;
      const v8h zero8 = {};
      pa.half8[0] = *(const v8h*)(pt + mloc * 16 + 8 * hi);  // K 0..15
      pa.half8[1] = zero8;                                    // K 16..31 pad
      for (int e = 0; e < 8; ++e)
        bv.h[e] = vh[(8 * hi + e) * 64 + wave * 16 + mloc];
      bv.half8[1] = zero8;
      acc = WMMA(pa.v, bv.v, acc);

#pragma unroll
      for (int b = 0; b < 3; ++b) {
#pragma unroll
        for (int r = 0; r < 4; ++r) {
          V16H ak;
          v8f cs;
#pragma unroll
          for (int e = 0; e < 8; ++e) {
            // A[m=d_row][k=l'] = kf[l'][16r+m], K padded 16->32 (transposed)
            ak.h[e] = kfh[b * 1024 + (8 * hi + e) * 64 + r * 16 + mloc];
          }
          ak.half8[1] = zero8;
#pragma unroll
          for (int j = 0; j < 8; ++j)
            cs[j] = S[(b * 64 + r * 16 + j + 8 * hi) * 64 + wave * 16 + mloc];
          v8f ds = WMMA(ak.v, bv.v, cs);
#pragma unroll
          for (int j = 0; j < 8; ++j)
            S[(b * 64 + r * 16 + j + 8 * hi) * 64 + wave * 16 + mloc] = ds[j];
        }
      }
    }

    // ---- denominator partials: thread t -> (l = t/8, d-group g = t%8) ----
    {
      const int l = t >> 3, g = t & 7;
      float part = 0.f;
      for (int b = 0; b < 3; ++b) {
        for (int dd = 0; dd < 8; ++dd) {
          const int d = g * 8 + dd;
          float cs = zf[b * 64 + d];
          for (int lp = 0; lp <= l; ++lp)
            cs += (float)kfh[b * 1024 + lp * 64 + d];
          part += (float)qfh[b * 1024 + l * 64 + d] * cs;
        }
      }
      denp[l * 8 + g] = part;
    }
    __syncthreads();
    if (t < 16) {
      float s = 0.f;
      for (int g = 0; g < 8; ++g) s += denp[t * 8 + g];
      den[t] = s;
    }
    if (t < 64) {  // advance running key-sums
      for (int b = 0; b < 3; ++b) {
        float s = 0.f;
        for (int lp = 0; lp < 16; ++lp) s += (float)kfh[b * 1024 + lp * 64 + t];
        zf[b * 64 + t] += s;
      }
    }
    __syncthreads();

    // ---- write output tile: out[n, l, h*64 + col] = num/den ----
#pragma unroll
    for (int j = 0; j < 8; ++j) {
      const int l = l0 + j + 8 * hi;
      const int col = h * 64 + wave * 16 + mloc;
      out[((size_t)(n * 1024 + l)) * 512 + col] = acc[j] / den[j + 8 * hi];
    }
    __syncthreads();  // before next chunk overwrites qfh/kfh/vh/pt
  }
}

// ---------------------------------------------------------------------------
extern "C" void kernel_launch(void* const* d_in, const int* in_sizes, int n_in,
                              void* d_out, int out_size, void* d_ws,
                              size_t ws_size, hipStream_t stream) {
  const float* query = (const float*)d_in[0];
  const float* key   = (const float*)d_in[1];
  const float* Wq    = (const float*)d_in[2];
  const float* Wk    = (const float*)d_in[3];
  const float* Wv    = (const float*)d_in[4];
  const float* posw  = (const float*)d_in[5];
  const float* posb  = (const float*)d_in[6];
  float* out = (float*)d_out;

  const size_t per = (size_t)2 * 8 * 1024 * 64;  // 1M floats each
  float* Qw = (float*)d_ws;
  float* Kw = Qw + per;
  float* Vw = Kw + per;

  fka_proj_kernel<<<3072, 128, 0, stream>>>(query, key, Wq, Wk, Wv, Qw, Kw, Vw);

  const size_t smem =
      (size_t)(3 * 64 * 64 + 3 * 64 + 64 + 64 + 16 + 16 * 8) * sizeof(float) +
      (size_t)(2 * 3 * 16 * 64 + 16 * 64 + 16 * 16) * sizeof(_Float16);
  hipFuncSetAttribute(reinterpret_cast<const void*>(fka_attn_kernel),
                      hipFuncAttributeMaxDynamicSharedMemorySize, (int)smem);
  fka_attn_kernel<<<16, 128, smem, stream>>>(Qw, Kw, Vw, posw, posb, out);
}
